// VaswaniMultiHeadAttention_42992622633336
// MI455X (gfx1250) — matrix operand order repaired and checked by reading, not compiled, not run on hardware
//
#include <hip/hip_runtime.h>
#include <hip/hip_bf16.h>

#define S_LEN 2048
#define EMB   1024
#define NH    16
#define DH    64
#define BATCH 2
#define SCALE_F 0.03125f  /* 1/sqrt(1024) */

typedef __bf16 v16bf  __attribute__((ext_vector_type(16)));
typedef __bf16 bf16x8 __attribute__((ext_vector_type(8)));
typedef __bf16 bf16x4 __attribute__((ext_vector_type(4)));
typedef float  v8f    __attribute__((ext_vector_type(8)));

static __device__ inline v16bf load_frag(const __bf16* p0, const __bf16* p1) {
  bf16x8 a = *(const bf16x8*)p0;   // 16B -> global_load_b128 / ds_load_b128
  bf16x8 b = *(const bf16x8*)p1;
  v16bf r;
#pragma unroll
  for (int i = 0; i < 8; ++i) { r[i] = a[i]; r[i + 8] = b[i]; }
  return r;
}

static __device__ inline v8f zero8() {
  v8f z = {0.f, 0.f, 0.f, 0.f, 0.f, 0.f, 0.f, 0.f};
  return z;
}

static __device__ inline v8f wmma_bf16(v16bf a, v16bf b, v8f c) {
  return __builtin_amdgcn_wmma_f32_16x16x32_bf16(false, a, false, b, (short)0, c,
                                                 false, false);
}

// ---------------- stage 0a: f32 -> bf16 elementwise (4 elems/thread) --------
__global__ void cvt_bf16_kernel(const float* __restrict__ src, __bf16* __restrict__ dst) {
  int i = (blockIdx.x * blockDim.x + threadIdx.x) * 4;
  float4 f = *(const float4*)(src + i);
  bf16x4 o;
  o[0] = (__bf16)f.x; o[1] = (__bf16)f.y; o[2] = (__bf16)f.z; o[3] = (__bf16)f.w;
  *(bf16x4*)(dst + i) = o;
}

// ---------------- stage 0b: W[K,N] f32 -> Wt[N,K] bf16 (LDS tiled) ----------
__global__ void transpose_bf16_kernel(const float* __restrict__ W, __bf16* __restrict__ Wt) {
  __shared__ __bf16 tile[32][33];
  const int bx = blockIdx.x * 32, by = blockIdx.y * 32;
  const int tx = threadIdx.x, ty = threadIdx.y;
#pragma unroll
  for (int j = 0; j < 32; j += 8)
    tile[ty + j][tx] = (__bf16)W[(by + ty + j) * EMB + bx + tx];
  __syncthreads();
#pragma unroll
  for (int j = 0; j < 32; j += 8)
    Wt[(bx + ty + j) * EMB + by + tx] = tile[tx][ty + j];
}

// ---------------- stage 1: projections  P = A * W^T + bias  (bf16 WMMA) -----
// 32x64 tile per wave, register double-buffered K loop (8 wmma / k-step).
// z=0: Qp = qb*Wq+bq ; z=1: Kp = vb*Wk+bk ; z=2: Vt[h][b][d][s] = vb*Wv+bv
__global__ void __launch_bounds__(256) proj_gemm_kernel(
    const __bf16* __restrict__ qb, const __bf16* __restrict__ vb,
    const __bf16* __restrict__ Wtq, const __bf16* __restrict__ Wtk, const __bf16* __restrict__ Wtv,
    const float* __restrict__ bq, const float* __restrict__ bk, const float* __restrict__ bv,
    __bf16* __restrict__ Qp, __bf16* __restrict__ Kp, __bf16* __restrict__ Vt) {
  const int z = blockIdx.z;
  const __bf16* A    = (z == 0) ? qb : vb;
  const __bf16* Wt   = (z == 0) ? Wtq : (z == 1) ? Wtk : Wtv;
  const float*  bias = (z == 0) ? bq  : (z == 1) ? bk  : bv;

  const int lane  = threadIdx.x & 31;
  const int wave  = blockIdx.x * 8 + (threadIdx.x >> 5);   // 2048 waves
  const int m_base = (wave & 127) * 32;                    // M = 4096, 32 rows/wave
  const int n_base = (wave >> 7) * 64;                     // N = 1024, 64 cols/wave
  const int nlo  = lane & 15;
  const int aoff = (lane & 16) ? 16 : 0;   // A-frag K half offset
  const int boff = (lane & 16) ? 16 : 0;   // B-frag K half offset

  const __bf16* arow0 = A + (m_base + nlo) * EMB + aoff;          // rows 0..15
  const __bf16* arow1 = arow0 + 16 * EMB;                         // rows 16..31
  const __bf16* brow0 = Wt + (n_base +  0 + nlo) * EMB + boff;
  const __bf16* brow1 = Wt + (n_base + 16 + nlo) * EMB + boff;
  const __bf16* brow2 = Wt + (n_base + 32 + nlo) * EMB + boff;
  const __bf16* brow3 = Wt + (n_base + 48 + nlo) * EMB + boff;

  v8f acc[8];
#pragma unroll
  for (int t = 0; t < 8; ++t) acc[t] = zero8();

  // prologue: fragments for k0 = 0
  v16bf a0 = load_frag(arow0, arow0 + 8);
  v16bf a1 = load_frag(arow1, arow1 + 8);
  v16bf b0 = load_frag(brow0, brow0 + 8);
  v16bf b1 = load_frag(brow1, brow1 + 8);
  v16bf b2 = load_frag(brow2, brow2 + 8);
  v16bf b3 = load_frag(brow3, brow3 + 8);

  for (int k0 = 0; k0 < EMB; k0 += 32) {
    const int kn = (k0 + 32) & (EMB - 1);   // wraps on last iter (valid addrs)
    // prefetch next k-step into fresh registers (overlaps with wmma below)
    v16bf na0 = load_frag(arow0 + kn, arow0 + kn + 8);
    v16bf na1 = load_frag(arow1 + kn, arow1 + kn + 8);
    v16bf nb0 = load_frag(brow0 + kn, brow0 + kn + 8);
    v16bf nb1 = load_frag(brow1 + kn, brow1 + kn + 8);
    v16bf nb2 = load_frag(brow2 + kn, brow2 + kn + 8);
    v16bf nb3 = load_frag(brow3 + kn, brow3 + kn + 8);

    acc[0] = wmma_bf16(a0, b0, acc[0]);
    acc[1] = wmma_bf16(a0, b1, acc[1]);
    acc[2] = wmma_bf16(a0, b2, acc[2]);
    acc[3] = wmma_bf16(a0, b3, acc[3]);
    acc[4] = wmma_bf16(a1, b0, acc[4]);
    acc[5] = wmma_bf16(a1, b1, acc[5]);
    acc[6] = wmma_bf16(a1, b2, acc[6]);
    acc[7] = wmma_bf16(a1, b3, acc[7]);

    a0 = na0; a1 = na1; b0 = nb0; b1 = nb1; b2 = nb2; b3 = nb3;
  }

#pragma unroll
  for (int mt = 0; mt < 2; ++mt) {
    const int mrow0 = m_base + mt * 16 + ((lane & 16) ? 8 : 0);
#pragma unroll
    for (int t = 0; t < 4; ++t) {
      const int n = n_base + t * 16 + nlo;
      const float bval = bias[n];
#pragma unroll
      for (int r = 0; r < 8; ++r) {
        const int m = mrow0 + r;
        const float val = acc[mt * 4 + t][r] + bval;
        if (z == 0) {
          Qp[m * EMB + n] = (__bf16)val;
        } else if (z == 1) {
          Kp[m * EMB + n] = (__bf16)val;
        } else {
          const int h = n >> 6, d = n & 63;
          const int bb = m >> 11, s = m & (S_LEN - 1);
          Vt[(((h * BATCH + bb) * DH + d) << 11) + s] = (__bf16)val;
        }
      }
    }
  }
}

// ---------------- stage 2: flash attention, one wave per (h,b,16 queries) ---
__global__ void __launch_bounds__(256) attn_kernel(
    const __bf16* __restrict__ Qp, const __bf16* __restrict__ Kp,
    const __bf16* __restrict__ Vt, float* __restrict__ out) {
  __shared__ __bf16 ldsP[8][16 * 32];   // per-wave P staging (C-layout -> A-layout)
  const int lane = threadIdx.x & 31;
  const int widx = threadIdx.x >> 5;
  const int wave = blockIdx.x * 8 + widx;       // 4096 waves
  const int qblk = wave & 127;
  const int hb   = wave >> 7;
  const int b = hb & 1, h = hb >> 1;

  const int nlo    = lane & 15;
  const int half8  = (lane & 16) ? 8 : 0;
  const int half16 = (lane & 16) ? 16 : 0;

  // Q fragments (16 rows x dh=64 -> two 16x32 A-frags), loaded once
  const int qrow = b * S_LEN + qblk * 16 + nlo;
  const __bf16* qp = Qp + qrow * EMB + h * DH + half16;
  const v16bf qf0 = load_frag(qp,      qp + 8);
  const v16bf qf1 = load_frag(qp + 32, qp + 40);

  float run_m[8], run_l[8];
#pragma unroll
  for (int r = 0; r < 8; ++r) { run_m[r] = -1e30f; run_l[r] = 0.f; }
  v8f o[4];
#pragma unroll
  for (int t = 0; t < 4; ++t) o[t] = zero8();

  __bf16* lp = ldsP[widx];
  const __bf16* vbase = Vt + (h * BATCH + b) * (DH * S_LEN);
  const __bf16* kbase = Kp + (b * S_LEN) * EMB + h * DH + half16;

  // prologue: K fragments for key block 0 (two 16-key tiles x two k-steps)
  const __bf16* kp0 = kbase + (0 + nlo) * EMB;
  const __bf16* kp1 = kbase + (16 + nlo) * EMB;
  v16bf kf00 = load_frag(kp0,      kp0 + 8);
  v16bf kf01 = load_frag(kp0 + 32, kp0 + 40);
  v16bf kf10 = load_frag(kp1,      kp1 + 8);
  v16bf kf11 = load_frag(kp1 + 32, kp1 + 40);

  for (int n0 = 0; n0 < S_LEN; n0 += 32) {
    const int nn = (n0 + 32) & (S_LEN - 1);   // wraps on last iter (valid addrs)

    // V fragments for CURRENT block, issued early: softmax VALU hides latency
    v16bf vf[4];
#pragma unroll
    for (int t = 0; t < 4; ++t) {
      const __bf16* vp = vbase + (t * 16 + nlo) * S_LEN + n0 + half16;
      vf[t] = load_frag(vp, vp + 8);
    }

    // prefetch NEXT block's K fragments
    const __bf16* nkp0 = kbase + (nn + nlo) * EMB;
    const __bf16* nkp1 = kbase + (nn + 16 + nlo) * EMB;
    v16bf nkf00 = load_frag(nkp0,      nkp0 + 8);
    v16bf nkf01 = load_frag(nkp0 + 32, nkp0 + 40);
    v16bf nkf10 = load_frag(nkp1,      nkp1 + 8);
    v16bf nkf11 = load_frag(nkp1 + 32, nkp1 + 40);

    // ---- scores: two 16x16 tiles over 32 keys (K fragments from prev iter)
    v8f sc[2];
    {
      v8f a = zero8();
      a = wmma_bf16(qf0, kf00, a);
      a = wmma_bf16(qf1, kf01, a);
      sc[0] = a;
      v8f c = zero8();
      c = wmma_bf16(qf0, kf10, c);
      c = wmma_bf16(qf1, kf11, c);
      sc[1] = c;
    }

    // ---- online softmax (per-row stats live in one 16-lane half)
    float corr[8];
#pragma unroll
    for (int r = 0; r < 8; ++r) {
      float s0 = sc[0][r] * SCALE_F;
      float s1 = sc[1][r] * SCALE_F;
      float mx = fmaxf(s0, s1);
      mx = fmaxf(mx, __shfl_xor(mx, 1, 32));
      mx = fmaxf(mx, __shfl_xor(mx, 2, 32));
      mx = fmaxf(mx, __shfl_xor(mx, 4, 32));
      mx = fmaxf(mx, __shfl_xor(mx, 8, 32));
      const float mnew = fmaxf(run_m[r], mx);
      const float c  = __expf(run_m[r] - mnew);
      const float p0 = __expf(s0 - mnew);
      const float p1 = __expf(s1 - mnew);
      float rs = p0 + p1;
      rs += __shfl_xor(rs, 1, 32);
      rs += __shfl_xor(rs, 2, 32);
      rs += __shfl_xor(rs, 4, 32);
      rs += __shfl_xor(rs, 8, 32);
      run_l[r] = run_l[r] * c + rs;
      run_m[r] = mnew;
      corr[r] = c;
      const int m = r + half8;                  // C-layout row
      lp[m * 32 + nlo]      = (__bf16)p0;       // keys [n0, n0+16)
      lp[m * 32 + 16 + nlo] = (__bf16)p1;       // keys [n0+16, n0+32)
    }

#pragma unroll
    for (int t = 0; t < 4; ++t)
#pragma unroll
      for (int r = 0; r < 8; ++r) o[t][r] *= corr[r];

    // fence DS stores before cross-lane DS fragment load (same wave)
    asm volatile("s_wait_dscnt 0x0" ::: "memory");

    // P as 16x32 A-fragment
    const __bf16* pp = lp + nlo * 32 + half16;
    v16bf pf = load_frag(pp, pp + 8);

    // ---- O += P(16x32) * V(32x64): 4 wmma, V fragments already in flight
#pragma unroll
    for (int t = 0; t < 4; ++t) o[t] = wmma_bf16(pf, vf[t], o[t]);

    kf00 = nkf00; kf01 = nkf01; kf10 = nkf10; kf11 = nkf11;
  }

#pragma unroll
  for (int r = 0; r < 8; ++r) run_l[r] = 1.0f / run_l[r];
#pragma unroll
  for (int t = 0; t < 4; ++t) {
#pragma unroll
    for (int r = 0; r < 8; ++r) {
      const int sq = qblk * 16 + r + half8;
      out[(b * S_LEN + sq) * EMB + h * DH + t * 16 + nlo] = o[t][r] * run_l[r];
    }
  }
}

// ---------------------------------------------------------------------------
extern "C" void kernel_launch(void* const* d_in, const int* in_sizes, int n_in,
                              void* d_out, int out_size, void* d_ws, size_t ws_size,
                              hipStream_t stream) {
  (void)in_sizes; (void)n_in; (void)out_size; (void)ws_size;
  const float* q  = (const float*)d_in[0];
  const float* v  = (const float*)d_in[1];
  const float* Wq = (const float*)d_in[2];
  const float* bq = (const float*)d_in[3];
  const float* Wk = (const float*)d_in[4];
  const float* bk = (const float*)d_in[5];
  const float* Wv = (const float*)d_in[6];
  const float* bv = (const float*)d_in[7];
  float* out = (float*)d_out;

  // workspace layout (bf16), total 46 MB
  char* ws = (char*)d_ws;
  const size_t MB = 1024ull * 1024ull;
  __bf16* qb  = (__bf16*)(ws);             //  8 MB  [4096,1024]
  __bf16* vb  = (__bf16*)(ws + 8 * MB);    //  8 MB
  __bf16* Wtq = (__bf16*)(ws + 16 * MB);   //  2 MB  [N,K]
  __bf16* Wtk = (__bf16*)(ws + 18 * MB);   //  2 MB
  __bf16* Wtv = (__bf16*)(ws + 20 * MB);   //  2 MB
  __bf16* Qp  = (__bf16*)(ws + 22 * MB);   //  8 MB  [4096,1024]
  __bf16* Kp  = (__bf16*)(ws + 30 * MB);   //  8 MB
  __bf16* Vt  = (__bf16*)(ws + 38 * MB);   //  8 MB  [H,B,dh,S]

  // stage 0: dtype conversion + weight transpose
  cvt_bf16_kernel<<<4096, 256, 0, stream>>>(q, qb);   // 4M elems, 4/thread
  cvt_bf16_kernel<<<4096, 256, 0, stream>>>(v, vb);
  dim3 tb(32, 8), tg(32, 32);
  transpose_bf16_kernel<<<tg, tb, 0, stream>>>(Wq, Wtq);
  transpose_bf16_kernel<<<tg, tb, 0, stream>>>(Wk, Wtk);
  transpose_bf16_kernel<<<tg, tb, 0, stream>>>(Wv, Wtv);

  // stage 1: three projection GEMMs (z selects Q/K/V), 32x64 tile per wave
  proj_gemm_kernel<<<dim3(256, 1, 3), 256, 0, stream>>>(
      qb, vb, Wtq, Wtk, Wtv, bq, bk, bv, Qp, Kp, Vt);

  // stage 2: streaming-softmax attention (pipelined K/V fetch)
  attn_kernel<<<512, 256, 0, stream>>>(Qp, Kp, Vt, out);
}
